// MambaBlock_17557826306470
// MI455X (gfx1250) — matrix operand order repaired and checked by reading, not compiled, not run on hardware
//
#include <hip/hip_runtime.h>
#include <hip/hip_bf16.h>

// ---------------- problem constants (from reference) ----------------
#define DM     1024          // d_model
#define DI     2048          // d_inner
#define NST    16            // d_state
#define DTRANK 64            // dt_rank
#define NB     2             // batch
#define LSEQ   2048          // seq len
#define BLROWS (NB * LSEQ)   // 4096 token rows

typedef __bf16 bf16_t;
typedef __attribute__((ext_vector_type(16))) __bf16 bf16x16_t;
typedef __attribute__((ext_vector_type(8)))  __bf16 bf16x8_t;
typedef __attribute__((ext_vector_type(8)))  float  f32x8_t;
typedef __attribute__((ext_vector_type(4)))  unsigned int u32x4_t;
typedef __attribute__((ext_vector_type(8)))  int  i32x8_t;
typedef __attribute__((ext_vector_type(4)))  int  i32x4_t;

// LDS A-panel geometry: rows of 64 bf16 (128 B) + 16 B TDM pad -> 144 B stride.
// 36*row mod 64 gives 16 distinct bank starts spaced 4 apart: conflict-free b128 reads.
#define PANEL_ROW_ELEMS 72           // 144 B / 2
#define PANEL_BYTES     (32 * 144)   // 4608 B per buffer

// ---------------- WMMA fragment helpers (wave32 layouts per CDNA5 ISA 7.12.2) ----
// A: 16x32 bf16 fragment. lane<16: K = {0..7, 16..23}; lane>=16: K = {8..15, 24..31}.
__device__ __forceinline__ bf16x16_t load_a_frag_lds(const bf16_t* S, int m0, int kk, int lane) {
    const int half = lane >> 4;
    const int row  = m0 + (lane & 15);
    const bf16_t* p = S + row * PANEL_ROW_ELEMS + kk + half * 8;
    bf16x8_t lo = *reinterpret_cast<const bf16x8_t*>(p);
    bf16x8_t hi = *reinterpret_cast<const bf16x8_t*>(p + 16);
    bf16x16_t a;
#pragma unroll
    for (int i = 0; i < 8; ++i) { a[i] = lo[i]; a[i + 8] = hi[i]; }
    return a;
}

__device__ __forceinline__ bf16x16_t load_a_frag(const bf16_t* __restrict__ A, int lda,
                                                 int m0, int k0, int lane) {
    const int half = lane >> 4;
    const int row  = m0 + (lane & 15);
    const bf16_t* p = A + (size_t)row * lda + k0 + half * 8;
    bf16x8_t lo = *reinterpret_cast<const bf16x8_t*>(p);
    bf16x8_t hi = *reinterpret_cast<const bf16x8_t*>(p + 16);
    bf16x16_t a;
#pragma unroll
    for (int i = 0; i < 8; ++i) { a[i] = lo[i]; a[i + 8] = hi[i]; }
    return a;
}

// B: 32x16 bf16 with B(k,n) = W[n,k], W row-major [N x K].
__device__ __forceinline__ bf16x16_t load_b_frag(const bf16_t* __restrict__ W, int ldw,
                                                 int n0, int k0, int lane) {
    const int half = lane >> 4;
    const int col  = n0 + (lane & 15);
    const bf16_t* p = W + (size_t)col * ldw + k0 + half * 8;
    bf16x8_t lo = *reinterpret_cast<const bf16x8_t*>(p);
    bf16x8_t hi = *reinterpret_cast<const bf16x8_t*>(p + 16);
    bf16x16_t b;
#pragma unroll
    for (int i = 0; i < 8; ++i) { b[i] = lo[i]; b[i + 8] = hi[i]; }
    return b;
}

__device__ __forceinline__ float softplusf(float x) {
    return (x > 20.f) ? x : log1pf(__expf(x));
}

// ---------------- TDM: stage a 32(rows) x 64(K) bf16 tile into LDS ----------------
// D# per CDNA5 ISA 8.3/8.4. pad_enable inserts 16 B after every 128 B row so the
// LDS panel lands with a 144 B row stride (bank-conflict-free fragment reads).
__device__ __forceinline__ void tdm_issue_panel(const bf16_t* gsrc, unsigned int lds_byte_off,
                                                int lda /*elements*/) {
    unsigned long long g = (unsigned long long)(uintptr_t)gsrc;
    u32x4_t g0;
    g0[0] = 1u;                                    // count=1, is_restore=0, gather off
    g0[1] = lds_byte_off;                          // LDS byte address
    g0[2] = (unsigned int)(g & 0xffffffffu);       // global_addr[31:0]
    g0[3] = (unsigned int)((g >> 32) & 0x01ffffffu) | (2u << 30);  // addr[56:32] | type=2

    unsigned int td0 = (unsigned int)lda;          // tensor dim0 (row length, elements)
    unsigned int td1 = 0x7fffffffu;                // tensor dim1: large, tiles are in-bounds
    unsigned long long s0 = (unsigned long long)lda;
    i32x8_t g1;
    g1[0] = (int)((1u << 16)                       // data_size = 1 -> 2-byte elements
                | (1u << 20)                       // pad_enable
                | (4u << 22)                       // pad_interval = 4 -> every 32 DWORDs (128 B)
                | (3u << 25));                     // pad_amount   = 3 -> 4 DWORDs (16 B)
    g1[1] = (int)((td0 & 0xffffu) << 16);          // [63:48] = tensor_dim0[15:0]
    g1[2] = (int)((td0 >> 16) | ((td1 & 0xffffu) << 16));
    g1[3] = (int)((td1 >> 16) | (64u << 16));      // [127:112] = tile_dim0 = 64
    g1[4] = (int)(32u);                            // tile_dim1 = 32, tile_dim2 = 0
    g1[5] = (int)(s0 & 0xffffffffu);               // tensor_dim0_stride[31:0]
    g1[6] = (int)((s0 >> 32) & 0xffffu);           // stride[47:32] | dim1_stride = 0
    g1[7] = 0;
    i32x4_t zero4 = {0, 0, 0, 0};
    i32x8_t zero8 = {0, 0, 0, 0, 0, 0, 0, 0};
    __builtin_amdgcn_tensor_load_to_lds(g0, g1, zero4, zero4, zero8, 0);
}

// ---------------- big GEMM: C[M x N] = A[M x K] * W[N x K]^T ----------------------
// Block = 8 waves. Block tile = 32(M) x 512(N); wave tile = 32 x 64 (2x4 WMMA tiles).
// A panel (32x64 per stage) staged in LDS via double-buffered TDM, shared by 8 waves.
// Requires: M % 32 == 0, N % 512 == 0, K % 64 == 0.
// MODE 0: plain f32 store. MODE 1: softplus(val + bias[n]) (dt path).
template <int MODE>
__global__ void __launch_bounds__(256)
gemm_wmma_tdm(const bf16_t* __restrict__ A, const bf16_t* __restrict__ W,
              float* __restrict__ C, const float* __restrict__ bias,
              int M, int N, int K, int lda, int ldw, int ldc) {
    __shared__ bf16_t sA[2][32 * PANEL_ROW_ELEMS];    // 2 x 4.5 KB padded panels
    const int lane  = threadIdx.x & 31;
    const int wave  = threadIdx.x >> 5;
    const int nb512 = N >> 9;
    const int m0    = (blockIdx.x / nb512) << 5;      // block M offset (32 rows)
    const int nwave = ((blockIdx.x % nb512) << 9) + (wave << 6);  // wave N offset

    f32x8_t acc[8] = {{}, {}, {}, {}, {}, {}, {}, {}};
    const unsigned int lds0 = (unsigned int)(uintptr_t)(void*)&sA[0][0];
    const unsigned int lds1 = (unsigned int)(uintptr_t)(void*)&sA[1][0];
    const bf16_t* Apanel = A + (size_t)m0 * lda;
    const int stages = K >> 6;

    if (threadIdx.x == 0) tdm_issue_panel(Apanel, lds0, lda);   // prologue: buf0

    for (int st = 0; st < stages; ++st) {
        if (threadIdx.x == 0) {
            if (st + 1 < stages) {
                // overlap: issue next panel into the other buffer, then cover this one
                tdm_issue_panel(Apanel + ((st + 1) << 6), ((st + 1) & 1) ? lds1 : lds0, lda);
                __builtin_amdgcn_s_wait_tensorcnt(1);
            } else {
                __builtin_amdgcn_s_wait_tensorcnt(0);
            }
        }
        __syncthreads();                               // publish buf[st&1]
        const bf16_t* S = &sA[st & 1][0];
        const int k0 = st << 6;
#pragma unroll
        for (int kk = 0; kk < 64; kk += 32) {
            bf16x16_t a0 = load_a_frag_lds(S, 0, kk, lane);
            bf16x16_t a1 = load_a_frag_lds(S, 16, kk, lane);
            bf16x16_t b[4];
#pragma unroll
            for (int s = 0; s < 4; ++s)
                b[s] = load_b_frag(W, ldw, nwave + (s << 4), k0 + kk, lane);
#pragma unroll
            for (int s = 0; s < 4; ++s) {
                acc[s]     = __builtin_amdgcn_wmma_f32_16x16x32_bf16(
                                 false, a0, false, b[s], (short)0, acc[s], false, false);
                acc[4 + s] = __builtin_amdgcn_wmma_f32_16x16x32_bf16(
                                 false, a1, false, b[s], (short)0, acc[4 + s], false, false);
            }
        }
        __syncthreads();                               // buf[st&1] free for stage st+2
    }

    const int ncol = lane & 15;
    const int half = lane >> 4;
#pragma unroll
    for (int mt = 0; mt < 2; ++mt) {
        const int mbase = m0 + (mt << 4) + (half << 3);  // D layout: M = r + 8*half
#pragma unroll
        for (int s = 0; s < 4; ++s) {
            const int n = nwave + (s << 4) + ncol;
            float bn = (MODE == 1) ? bias[n] : 0.f;
#pragma unroll
            for (int r = 0; r < 8; ++r) {
                float v = acc[mt * 4 + s][r];
                if (MODE == 1) v = softplusf(v + bn);
                C[(size_t)(mbase + r) * ldc + n] = v;
            }
        }
    }
}

// ---------------- small-N GEMM (x_proj, N=96): direct loads, branch-free ---------
template <int NT>
__global__ void __launch_bounds__(256)
gemm_wmma_direct(const bf16_t* __restrict__ A, const bf16_t* __restrict__ W,
                 float* __restrict__ C, int M, int N, int K,
                 int lda, int ldw, int ldc) {
    const int lane   = threadIdx.x & 31;
    const int wave   = threadIdx.x >> 5;
    const int tilesN = N / (16 * NT);
    const int wid    = blockIdx.x * 8 + wave;
    if (wid >= (M >> 4) * tilesN) return;
    const int m0    = (wid / tilesN) << 4;
    const int nbase = (wid % tilesN) * (16 * NT);

    f32x8_t acc[NT];
#pragma unroll
    for (int s = 0; s < NT; ++s) acc[s] = f32x8_t{};

    for (int k0 = 0; k0 < K; k0 += 32) {
        bf16x16_t a = load_a_frag(A, lda, m0, k0, lane);
#pragma unroll
        for (int s = 0; s < NT; ++s) {
            bf16x16_t b = load_b_frag(W, ldw, nbase + (s << 4), k0, lane);
            acc[s] = __builtin_amdgcn_wmma_f32_16x16x32_bf16(
                         false, a, false, b, (short)0, acc[s], false, false);
        }
    }

    const int ncol  = lane & 15;
    const int mbase = m0 + ((lane >> 4) << 3);
#pragma unroll
    for (int s = 0; s < NT; ++s) {
        const int n = nbase + (s << 4) + ncol;
#pragma unroll
        for (int r = 0; r < 8; ++r)
            C[(size_t)(mbase + r) * ldc + n] = acc[s][r];
    }
}

// ---------------- elementwise casts ----------------
__global__ void cast_f32_to_bf16(const float* __restrict__ in, bf16_t* __restrict__ out, int n) {
    int i = blockIdx.x * 256 + threadIdx.x;
    if (i < n) out[i] = (bf16_t)in[i];
}

// dt_low = x_dbl[:, 0:64] -> bf16 [BLROWS x 64]
__global__ void extract_dtlow_bf16(const float* __restrict__ xdbl, bf16_t* __restrict__ out) {
    int i = blockIdx.x * 256 + threadIdx.x;
    if (i < BLROWS * DTRANK) {
        int r = i >> 6, c = i & 63;
        out[i] = (bf16_t)xdbl[r * (DTRANK + 2 * NST) + c];
    }
}

// ---------------- depthwise causal conv1d + bias + SiLU ----------------
__global__ void conv_bias_silu(const float* __restrict__ xz, const float* __restrict__ cw,
                               const float* __restrict__ cb, float* __restrict__ xact,
                               bf16_t* __restrict__ xact_bf) {
    int idx = blockIdx.x * 256 + threadIdx.x;    // over NB*LSEQ*DI
    if (idx >= BLROWS * DI) return;
    int d  = idx % DI;
    int bl = idx / DI;
    int l  = bl % LSEQ;
    float acc = cb[d];
#pragma unroll
    for (int j = 0; j < 4; ++j) {
        int ls = l - 3 + j;
        if (ls >= 0) acc = fmaf(cw[d * 4 + j], xz[(size_t)(bl - 3 + j) * (2 * DI) + d], acc);
    }
    float s = acc / (1.f + __expf(-acc));        // SiLU
    xact[idx]    = s;
    xact_bf[idx] = (bf16_t)s;
}

// ---------------- selective scan: one lane per (b, d) channel ----------------
__global__ void __launch_bounds__(256)
ssm_scan(const float* __restrict__ delta, const float* __restrict__ xdbl,
         const float* __restrict__ xact, const float* __restrict__ xz,
         const float* __restrict__ A_log, const float* __restrict__ Dskip,
         bf16_t* __restrict__ ybf) {
    const int d = blockIdx.x * 256 + threadIdx.x;
    const int b = blockIdx.y;
    __shared__ float sBC[32];

    float h[NST], Ad[NST];
#pragma unroll
    for (int n = 0; n < NST; ++n) {
        h[n]  = 0.f;
        Ad[n] = -__expf(A_log[d * NST + n]);     // A = -exp(A_log)
    }
    const float Dd = Dskip[d];

    for (int t = 0; t < LSEQ; ++t) {
        const int row = b * LSEQ + t;
        __syncthreads();
        if (threadIdx.x < 32)
            sBC[threadIdx.x] = xdbl[(size_t)row * (DTRANK + 2 * NST) + DTRANK + threadIdx.x];
        __syncthreads();

        const float dt = delta[(size_t)row * DI + d];
        const float u  = xact [(size_t)row * DI + d];
        const float du = dt * u;
        float y = 0.f;
#pragma unroll
        for (int n = 0; n < NST; ++n) {
            h[n] = fmaf(__expf(dt * Ad[n]), h[n], du * sBC[n]);   // h = dA*h + dBu
            y    = fmaf(h[n], sBC[16 + n], y);                    // y += h*C
        }
        y = fmaf(u, Dd, y);                                       // + D skip
        const float z = xz[(size_t)row * (2 * DI) + DI + d];      // gate input
        y *= z / (1.f + __expf(-z));                              // * silu(z)
        ybf[(size_t)row * DI + d] = (bf16_t)y;
    }
}

// ---------------- host-side orchestration ----------------
static inline int ceil_div(int a, int b) { return (a + b - 1) / b; }

extern "C" void kernel_launch(void* const* d_in, const int* in_sizes, int n_in,
                              void* d_out, int out_size, void* d_ws, size_t ws_size,
                              hipStream_t stream) {
    const float* x          = (const float*)d_in[0];   // (2,2048,1024)
    const float* in_proj_w  = (const float*)d_in[1];   // (4096,1024)
    const float* conv_w     = (const float*)d_in[2];   // (2048,1,4)
    const float* conv_b     = (const float*)d_in[3];   // (2048,)
    const float* x_proj_w   = (const float*)d_in[4];   // (96,2048)
    const float* dt_proj_w  = (const float*)d_in[5];   // (2048,64)
    const float* dt_proj_b  = (const float*)d_in[6];   // (2048,)
    const float* A_log      = (const float*)d_in[7];   // (2048,16)
    const float* D_skip     = (const float*)d_in[8];   // (2048,)
    const float* out_proj_w = (const float*)d_in[9];   // (1024,2048)
    float* out = (float*)d_out;                        // (2,2048,1024)

    char* ws = (char*)d_ws;
    bf16_t* x_bf      = (bf16_t*)(ws + 0);             //  8 MB  [4096x1024]
    bf16_t* winb      = (bf16_t*)(ws + 8388608);       //  8 MB  [4096x1024]
    float*  xz        = (float*) (ws + 16777216);      // 64 MB  [4096x4096]
    float*  xact      = (float*) (ws + 83886080);      // 32 MB  [4096x2048]
    bf16_t* xact_bf   = (bf16_t*)(ws + 117440512);     // 16 MB
    bf16_t* xprojw_bf = (bf16_t*)(ws + 134217728);     // 384 KB [96x2048]
    float*  xdbl      = (float*) (ws + 134610944);     // 1.5 MB [4096x96]
    bf16_t* dtlow_bf  = (bf16_t*)(ws + 136183808);     // 512 KB [4096x64]
    bf16_t* dtw_bf    = (bf16_t*)(ws + 136708096);     // 256 KB [2048x64]
    float*  delta     = (float*) (ws + 136970240);     // 32 MB  [4096x2048]
    bf16_t* y_bf      = (bf16_t*)(ws + 170524672);     // 16 MB  [4096x2048]
    bf16_t* outw_bf   = (bf16_t*)(ws + 187301888);     //  4 MB  [1024x2048]

    // 1) casts to bf16
    cast_f32_to_bf16<<<ceil_div(BLROWS * DM, 256), 256, 0, stream>>>(x, x_bf, BLROWS * DM);
    cast_f32_to_bf16<<<ceil_div(2 * DI * DM, 256), 256, 0, stream>>>(in_proj_w, winb, 2 * DI * DM);
    cast_f32_to_bf16<<<ceil_div(96 * DI, 256), 256, 0, stream>>>(x_proj_w, xprojw_bf, 96 * DI);
    cast_f32_to_bf16<<<ceil_div(DI * DTRANK, 256), 256, 0, stream>>>(dt_proj_w, dtw_bf, DI * DTRANK);
    cast_f32_to_bf16<<<ceil_div(DM * DI, 256), 256, 0, stream>>>(out_proj_w, outw_bf, DM * DI);

    // 2) in_proj GEMM: xz[4096 x 4096] = x * in_proj_w^T   (double-buffered TDM A)
    {
        int M = BLROWS, N = 2 * DI, K = DM;
        int blocks = (M / 32) * (N / 512);             // 128 * 8 = 1024
        gemm_wmma_tdm<0><<<blocks, 256, 0, stream>>>(x_bf, winb, xz, nullptr,
                                                     M, N, K, K, K, N);
    }

    // 3) depthwise causal conv + bias + SiLU
    conv_bias_silu<<<ceil_div(BLROWS * DI, 256), 256, 0, stream>>>(xz, conv_w, conv_b,
                                                                   xact, xact_bf);

    // 4) x_proj GEMM: x_dbl[4096 x 96] = x_act * x_proj_w^T (direct, 32-col waves)
    {
        int M = BLROWS, N = 96, K = DI;
        int wids = (M / 16) * (N / 32);                // 256 * 3 = 768
        gemm_wmma_direct<2><<<ceil_div(wids, 8), 256, 0, stream>>>(xact_bf, xprojw_bf, xdbl,
                                                                   M, N, K, K, K, N);
    }

    // 5) dt_low -> bf16, then dt GEMM + fused bias+softplus: delta[4096 x 2048]
    extract_dtlow_bf16<<<ceil_div(BLROWS * DTRANK, 256), 256, 0, stream>>>(xdbl, dtlow_bf);
    {
        int M = BLROWS, N = DI, K = DTRANK;            // K = 64 -> single TDM stage
        int blocks = (M / 32) * (N / 512);             // 128 * 4 = 512
        gemm_wmma_tdm<1><<<blocks, 256, 0, stream>>>(dtlow_bf, dtw_bf, delta, dt_proj_b,
                                                     M, N, K, K, K, N);
    }

    // 6) selective scan + skip + gate (writes gated y as bf16)
    {
        dim3 grid(DI / 256, NB);
        ssm_scan<<<grid, 256, 0, stream>>>(delta, xdbl, xact, xz, A_log, D_skip, y_bf);
    }

    // 7) out_proj GEMM: out[4096 x 1024] = y * out_proj_w^T  (f32 to d_out)
    {
        int M = BLROWS, N = DM, K = DI;
        int blocks = (M / 32) * (N / 512);             // 128 * 2 = 256
        gemm_wmma_tdm<0><<<blocks, 256, 0, stream>>>(y_bf, outw_bf, out, nullptr,
                                                     M, N, K, K, K, N);
    }
}
